// TransformerBlock_31224412242052
// MI455X (gfx1250) — matrix operand order repaired and checked by reading, not compiled, not run on hardware
//
#include <hip/hip_runtime.h>
#include <cmath>

// ---------------------------------------------------------------------------
// Transformer block for MI455X (gfx1250): bf16 WMMA with LDS tile sharing via
// the CDNA5 async Global->LDS path (ASYNCcnt), double-buffered. Flash-style
// attention with gaussian reweight folded into softmax as a log-bias.
// ---------------------------------------------------------------------------

typedef __attribute__((ext_vector_type(16))) __bf16 v16bf;
typedef __attribute__((ext_vector_type(8)))  float  v8f;

#define Bn   4
#define Sn   1024
#define Dn   1024
#define Hn   16
#define DHn  64
#define Fn   4096
#define Mtot (Bn * Sn)   /* 4096 token rows */

#define WAIT_ASYNC_0 asm volatile("s_wait_asynccnt 0" ::: "memory")
#define WAIT_ASYNC_1 asm volatile("s_wait_asynccnt 1" ::: "memory")
#define WAIT_ASYNC_2 asm volatile("s_wait_asynccnt 2" ::: "memory")

// 16-byte per-lane Global -> LDS async copy (GLOBAL_LOAD_ASYNC_TO_LDS_B128,
// tracked by ASYNCcnt). VDST = per-lane LDS byte address, VADDR = 64-bit
// global address in a VGPR pair, SADDR = off.
__device__ __forceinline__ void async_copy16(const __bf16* gsrc, __bf16* ldst) {
    unsigned int       l = (unsigned int)(unsigned long long)ldst;
    unsigned long long g = (unsigned long long)gsrc;
    asm volatile("global_load_async_to_lds_b128 %0, %1, off"
                 :: "v"(l), "v"(g) : "memory");
}

// ---------------------------------------------------------------------------
// Fragment loaders (wave32 layouts from CDNA5 ISA 7.12.2)
// ---------------------------------------------------------------------------

// A fragment from a precomputed per-lane pointer (row*lda + kbase already
// applied): dwords 0-3 = K[kb..kb+7], dwords 4-7 = K[kb+16..kb+23].
__device__ __forceinline__ v16bf frag_a_load_ptr(const __bf16* p) {
    union { uint4 q[2]; v16bf v; } f;
    f.q[0] = *(const uint4*)(p);
    f.q[1] = *(const uint4*)(p + 16);
    return f.v;
}

// A (16x32 bf16, row-major, leading dim lda): lane l: row=l&15, kb=(l>>4)*8
__device__ __forceinline__ v16bf frag_a_load(const __bf16* A, int lda) {
    int lane = threadIdx.x & 31;
    return frag_a_load_ptr(A + (lane & 15) * lda + ((lane >> 4) << 3));
}

// B (32x16 bf16) from an N-major ("Bt") buffer, leading dim ldb:
// lane l: col=l&15, kb=(l>>4)*8 ; dwords 0-3 = K[kb..kb+7], dwords 4-7 = K[kb+16..kb+23]
__device__ __forceinline__ v16bf frag_b_load(const __bf16* Bt, int ldb) {
    int lane = threadIdx.x & 31;
    int col  = lane & 15;
    int kb   = (lane >> 4) << 3;
    const __bf16* p = Bt + col * ldb + kb;
    union { uint4 q[2]; v16bf v; } f;
    f.q[0] = *(const uint4*)(p);
    f.q[1] = *(const uint4*)(p + 16);
    return f.v;
}

__device__ __forceinline__ v8f wmma_bf16(v16bf a, v16bf b, v8f c) {
    return __builtin_amdgcn_wmma_f32_16x16x32_bf16(
        false, a, false, b, (short)0, c, false, false);
}

// ---------------------------------------------------------------------------
// Packing kernels
// ---------------------------------------------------------------------------

__global__ void f32_to_bf16_kernel(const float* __restrict__ in,
                                   __bf16* __restrict__ out, int n) {
    int i = blockIdx.x * 256 + threadIdx.x;
    if (i < n) out[i] = (__bf16)in[i];
}

// in: [K][N] f32  ->  out: [N][K] bf16
__global__ void transpose_bf16_kernel(const float* __restrict__ in,
                                      __bf16* __restrict__ out, int K, int N) {
    int i = blockIdx.x * 256 + threadIdx.x;
    if (i < K * N) {
        int k = i / N, n = i % N;
        out[(size_t)n * K + k] = (__bf16)in[i];
    }
}

// softmax additive bias: mask==0 -> -inf, else log(gauss_weight + 1e-10)
__global__ void attn_bias_kernel(const int* __restrict__ mask,
                                 const float* __restrict__ gw,
                                 float* __restrict__ bias, int n) {
    int i = blockIdx.x * 256 + threadIdx.x;
    if (i < n) bias[i] = mask[i] ? logf(gw[i] + 1e-10f) : -3.0e38f;
}

// ---------------------------------------------------------------------------
// Unified bf16 GEMM: block tile 128(M) x 64(N), 8 waves stacked along M share
// a double-buffered B tile (64 cols x 32 k, 4KB) staged in LDS via async copy.
// Steady-state loop is branch-free (last K-step peeled).
// Epilogue modes:
//   0: f32 out = A@Bt + bias
//   1: bf16 out = gelu(A@Bt + bias)
//   2: Q scatter  [B,H,S,DH] * 1/sqrt(DH)
//   3: K scatter  [B,H,S,DH]
//   4: V^T scatter [B,H,DH,S]
// ---------------------------------------------------------------------------

__global__ __launch_bounds__(256)
void gemm_bf16_kernel(const __bf16* __restrict__ A, const __bf16* __restrict__ Bt,
                      const float* __restrict__ bias,
                      float* __restrict__ outF, __bf16* __restrict__ outH,
                      int M, int N, int K, int mode) {
    __shared__ __align__(16) __bf16 btile[2][64 * 32];

    int tilesN = N >> 6;
    int bm = blockIdx.x / tilesN;
    int tn = blockIdx.x % tilesN;
    int w = threadIdx.x >> 5, lane = threadIdx.x & 31;
    int tm = bm * 8 + w;
    int t = threadIdx.x;

    // per-thread B-tile copy coordinates: 256 threads x 16B = 4KB tile
    int bc = t >> 2;              // col within 64-wide strip
    int bk = (t & 3) << 3;        // k offset 0,8,16,24
    const __bf16* gB = Bt + (size_t)(tn * 64 + bc) * K + bk;
    __bf16* lB0 = &btile[0][t * 8];
    __bf16* lB1 = &btile[1][t * 8];

    async_copy16(gB, lB0);        // stage 0

    // per-lane A pointer, advanced by 32 elements per K-step
    const __bf16* pA = A + (size_t)tm * 16 * K
                         + (size_t)(lane & 15) * K + ((lane >> 4) << 3);

    const v8f z = {0.f,0.f,0.f,0.f,0.f,0.f,0.f,0.f};
    v8f c[4] = {z, z, z, z};

    auto compute_step = [&](int cur) {
        v16bf a = frag_a_load_ptr(pA);
        __builtin_prefetch((const void*)(pA + 256), 0, 1);
        pA += 32;
        const __bf16* bb = &btile[cur][0];
#pragma unroll
        for (int j = 0; j < 4; ++j)
            c[j] = wmma_bf16(a, frag_b_load(bb + (size_t)(j * 16) * 32, 32), c[j]);
    };

    int k0 = 0;
    for (; k0 < K - 32; k0 += 32) {          // steady state: branch-free body
        int cur = (k0 >> 5) & 1;
        async_copy16(gB + k0 + 32, cur ? lB0 : lB1);
        WAIT_ASYNC_1;
        __syncthreads();
        compute_step(cur);
        __syncthreads();
    }
    WAIT_ASYNC_0;                             // peeled last step
    __syncthreads();
    compute_step((k0 >> 5) & 1);

    int nl = lane & 15, mb = (lane >> 4) << 3;
#pragma unroll
    for (int j = 0; j < 4; ++j) {
#pragma unroll
        for (int r = 0; r < 8; ++r) {
            int row = tm * 16 + mb + r;
            int col = tn * 64 + j * 16 + nl;
            float v = c[j][r] + bias[col];
            if (mode == 0) {
                outF[(size_t)row * N + col] = v;
            } else if (mode == 1) {
                float g = 0.5f * v * (1.0f + erff(v * 0.70710678118654752f));
                outH[(size_t)row * N + col] = (__bf16)g;
            } else {
                int b = row >> 10, s = row & (Sn - 1);
                int h = col >> 6,  dh = col & (DHn - 1);
                if (mode == 2)
                    outH[((size_t)(b * Hn + h) * Sn + s) * DHn + dh] = (__bf16)(v * 0.125f);
                else if (mode == 3)
                    outH[((size_t)(b * Hn + h) * Sn + s) * DHn + dh] = (__bf16)v;
                else
                    outH[((size_t)(b * Hn + h) * DHn + dh) * Sn + s] = (__bf16)v;
            }
        }
    }
}

// ---------------------------------------------------------------------------
// Flash attention: block = one (b,h) slice x 8 query tiles (one per wave).
// K tile (32 keys x 64 dh) and V^T tile (64 dh x 32 keys) are shared by all 8
// waves, staged via async copies, double-buffered, last step peeled. Online
// softmax with additive key bias. P round-trips through per-wave LDS.
// ---------------------------------------------------------------------------

__global__ __launch_bounds__(256)
void attn_kernel(const __bf16* __restrict__ qb, const __bf16* __restrict__ kb,
                 const __bf16* __restrict__ vt, const float* __restrict__ biasb,
                 __bf16* __restrict__ ctx) {
    __shared__ __align__(16) __bf16 ktile[2][32 * 64];   // [key][dh]
    __shared__ __align__(16) __bf16 vtile[2][64 * 32];   // [dh][key]
    __shared__ __align__(16) __bf16 ptile[8][16][32];

    int w = threadIdx.x >> 5, lane = threadIdx.x & 31, t = threadIdx.x;
    int qg = blockIdx.x & 7;          // 8 blocks per (b,h)
    int bh = blockIdx.x >> 3;
    int h = bh & (Hn - 1), b = bh >> 4;
    int qt = qg * 8 + w;              // this wave's 16-row query tile

    const __bf16* Q  = qb + (((size_t)(b * Hn + h) * Sn) + qt * 16) * DHn;
    const __bf16* Kp = kb + ((size_t)(b * Hn + h) * Sn) * DHn;   // [key][dh]
    const __bf16* Vp = vt + ((size_t)(b * Hn + h) * DHn) * Sn;   // [dh][key]
    const float* bias = biasb + b * Sn;

    // per-thread copy coordinates
    int kr = t >> 3, kc = (t & 7) << 3;   // ktile: key row, dh offset
    int vr = t >> 2, vc = (t & 3) << 3;   // vtile: dh row, key offset
    const __bf16* gK = Kp + (size_t)kr * DHn + kc;
    const __bf16* gV = Vp + (size_t)vr * Sn + vc;

    async_copy16(gK, &ktile[0][t * 8]);   // stage 0
    async_copy16(gV, &vtile[0][t * 8]);

    v16bf aq0 = frag_a_load(Q, DHn);        // dh 0..31
    v16bf aq1 = frag_a_load(Q + 32, DHn);   // dh 32..63

    const v8f z = {0.f,0.f,0.f,0.f,0.f,0.f,0.f,0.f};
    v8f acc[4] = {z, z, z, z};
    float mrun[8], lrun[8];
#pragma unroll
    for (int r = 0; r < 8; ++r) { mrun[r] = -3.0e38f; lrun[r] = 0.f; }

    int nl = lane & 15;
    int mb = (lane >> 4) << 3;

    auto attn_step = [&](int k0, int cur) {
        const __bf16* kt  = &ktile[cur][0];
        const __bf16* vtp = &vtile[cur][0];

        // scores: two 16x16 tiles (keys +0..15, +16..31), K-dim = DH = 64
        v8f s0 = z, s1 = z;
        s0 = wmma_bf16(aq0, frag_b_load(kt, DHn), s0);
        s0 = wmma_bf16(aq1, frag_b_load(kt + 32, DHn), s0);
        s1 = wmma_bf16(aq0, frag_b_load(kt + 16 * DHn, DHn), s1);
        s1 = wmma_bf16(aq1, frag_b_load(kt + 16 * DHn + 32, DHn), s1);

        float bb0 = bias[k0 + nl];
        float bb1 = bias[k0 + 16 + nl];

        float p0[8], p1[8], scl[8];
#pragma unroll
        for (int r = 0; r < 8; ++r) {
            float t0 = s0[r] + bb0;
            float t1 = s1[r] + bb1;
            float mx = fmaxf(t0, t1);
#pragma unroll
            for (int d = 1; d < 16; d <<= 1)
                mx = fmaxf(mx, __shfl_xor(mx, d, 32));
            float mn = fmaxf(mrun[r], mx);
            bool dead = (mn <= -1.0e30f);
            float sc = dead ? 0.f : __expf(mrun[r] - mn);
            float e0 = dead ? 0.f : __expf(t0 - mn);
            float e1 = dead ? 0.f : __expf(t1 - mn);
            float rs = e0 + e1;
#pragma unroll
            for (int d = 1; d < 16; d <<= 1)
                rs += __shfl_xor(rs, d, 32);
            lrun[r] = lrun[r] * sc + rs;
            mrun[r] = mn;
            scl[r] = sc; p0[r] = e0; p1[r] = e1;
        }

#pragma unroll
        for (int j = 0; j < 4; ++j)
#pragma unroll
            for (int r = 0; r < 8; ++r) acc[j][r] *= scl[r];

        // P (16x32) C-layout -> per-wave LDS -> A-fragment
#pragma unroll
        for (int r = 0; r < 8; ++r) {
            ptile[w][mb + r][nl]      = (__bf16)p0[r];
            ptile[w][mb + r][nl + 16] = (__bf16)p1[r];
        }
        asm volatile("s_wait_dscnt 0" ::: "memory");  // DS in-order within wave
        v16bf pa = frag_a_load(&ptile[w][0][0], 32);

        // ctx += P @ V  (K = 32 keys, four 16-wide dh tiles)
#pragma unroll
        for (int j = 0; j < 4; ++j)
            acc[j] = wmma_bf16(pa, frag_b_load(vtp + (size_t)(j * 16) * 32, 32), acc[j]);
    };

    int k0 = 0;
    for (; k0 < Sn - 32; k0 += 32) {          // steady state: branch-free body
        int cur = (k0 >> 5) & 1;
        async_copy16(gK + (size_t)(k0 + 32) * DHn, &ktile[cur ^ 1][t * 8]);
        async_copy16(gV + (k0 + 32),               &vtile[cur ^ 1][t * 8]);
        WAIT_ASYNC_2;
        __syncthreads();
        attn_step(k0, cur);
        __syncthreads();
    }
    WAIT_ASYNC_0;                              // peeled last step
    __syncthreads();
    attn_step(k0, (k0 >> 5) & 1);

    float inv[8];
#pragma unroll
    for (int r = 0; r < 8; ++r) inv[r] = (lrun[r] > 0.f) ? (1.0f / lrun[r]) : 0.f;

#pragma unroll
    for (int j = 0; j < 4; ++j)
#pragma unroll
        for (int r = 0; r < 8; ++r) {
            int row = b * Sn + qt * 16 + mb + r;
            int col = h * DHn + j * 16 + nl;
            ctx[(size_t)row * Dn + col] = (__bf16)(acc[j][r] * inv[r]);
        }
}

// ---------------------------------------------------------------------------
// residual add + LayerNorm over DIM=1024 (one block per row, 4 elems/thread)
// ---------------------------------------------------------------------------

__global__ __launch_bounds__(256)
void add_ln_kernel(const float* __restrict__ a, const float* __restrict__ res,
                   const float* __restrict__ g, const float* __restrict__ be,
                   float* __restrict__ outF, __bf16* __restrict__ outH) {
    __shared__ float red[256];
    int row = blockIdx.x, t = threadIdx.x;
    const float* pa = a + (size_t)row * Dn;
    const float* pr = res + (size_t)row * Dn;

    float v[4], s = 0.f;
#pragma unroll
    for (int i = 0; i < 4; ++i) {
        int idx = t + i * 256;
        v[i] = pa[idx] + pr[idx];
        s += v[i];
    }
    red[t] = s; __syncthreads();
    for (int off = 128; off > 0; off >>= 1) {
        if (t < off) red[t] += red[t + off];
        __syncthreads();
    }
    float mean = red[0] * (1.0f / Dn);
    __syncthreads();

    float s2 = 0.f;
#pragma unroll
    for (int i = 0; i < 4; ++i) { float d = v[i] - mean; s2 += d * d; }
    red[t] = s2; __syncthreads();
    for (int off = 128; off > 0; off >>= 1) {
        if (t < off) red[t] += red[t + off];
        __syncthreads();
    }
    float var = red[0] * (1.0f / Dn);
    float rinv = rsqrtf(var + 1e-12f);

#pragma unroll
    for (int i = 0; i < 4; ++i) {
        int idx = t + i * 256;
        float o = (v[i] - mean) * rinv * g[idx] + be[idx];
        outF[(size_t)row * Dn + idx] = o;
        if (outH) outH[(size_t)row * Dn + idx] = (__bf16)o;
    }
}

// ---------------------------------------------------------------------------
// Host launcher
// ---------------------------------------------------------------------------

extern "C" void kernel_launch(void* const* d_in, const int* in_sizes, int n_in,
                              void* d_out, int out_size, void* d_ws, size_t ws_size,
                              hipStream_t stream) {
    const float* x    = (const float*)d_in[0];
    const int*   msk  = (const int*)d_in[1];
    const float* gw   = (const float*)d_in[2];
    const float* Wq   = (const float*)d_in[3];
    const float* bq   = (const float*)d_in[4];
    const float* Wk   = (const float*)d_in[5];
    const float* bk   = (const float*)d_in[6];
    const float* Wv   = (const float*)d_in[7];
    const float* bv   = (const float*)d_in[8];
    const float* Wo   = (const float*)d_in[9];
    const float* bo   = (const float*)d_in[10];
    const float* ln1g = (const float*)d_in[11];
    const float* ln1b = (const float*)d_in[12];
    const float* W1   = (const float*)d_in[13];
    const float* b1   = (const float*)d_in[14];
    const float* W2   = (const float*)d_in[15];
    const float* b2   = (const float*)d_in[16];
    const float* ln2g = (const float*)d_in[17];
    const float* ln2b = (const float*)d_in[18];
    float* out = (float*)d_out;

    char* ws = (char*)d_ws;
    size_t off = 0;
    auto alloc = [&](size_t bytes) -> void* {
        void* p = ws + off;
        off += (bytes + 255) & ~(size_t)255;
        return p;
    };

    const size_t TOK = (size_t)Mtot * Dn;      // 4M elements
    __bf16* xbf  = (__bf16*)alloc(TOK * 2);
    __bf16* wqt  = (__bf16*)alloc((size_t)Dn * Dn * 2);
    __bf16* wkt  = (__bf16*)alloc((size_t)Dn * Dn * 2);
    __bf16* wvt  = (__bf16*)alloc((size_t)Dn * Dn * 2);
    __bf16* wot  = (__bf16*)alloc((size_t)Dn * Dn * 2);
    __bf16* w1t  = (__bf16*)alloc((size_t)Dn * Fn * 2);
    __bf16* w2t  = (__bf16*)alloc((size_t)Fn * Dn * 2);
    __bf16* qbuf = (__bf16*)alloc(TOK * 2);
    __bf16* kbuf = (__bf16*)alloc(TOK * 2);
    __bf16* vtbf = (__bf16*)alloc(TOK * 2);
    __bf16* ctxb = (__bf16*)alloc(TOK * 2);
    float*  attb = (float*)alloc((size_t)Bn * Sn * 4);
    float*  sapre= (float*)alloc(TOK * 4);
    float*  saf  = (float*)alloc(TOK * 4);
    __bf16* sabf = (__bf16*)alloc(TOK * 2);
    __bf16* hbf  = (__bf16*)alloc((size_t)Mtot * Fn * 2);
    float*  ffnf = (float*)alloc(TOK * 4);

    dim3 blk(256);
    dim3 g512(512), g2048(2048);

    // 1) pack activations + weights to bf16 (weights transposed to N-major)
    f32_to_bf16_kernel<<<dim3((TOK + 255) / 256), blk, 0, stream>>>(x, xbf, (int)TOK);
    transpose_bf16_kernel<<<dim3((Dn * Dn + 255) / 256), blk, 0, stream>>>(Wq, wqt, Dn, Dn);
    transpose_bf16_kernel<<<dim3((Dn * Dn + 255) / 256), blk, 0, stream>>>(Wk, wkt, Dn, Dn);
    transpose_bf16_kernel<<<dim3((Dn * Dn + 255) / 256), blk, 0, stream>>>(Wv, wvt, Dn, Dn);
    transpose_bf16_kernel<<<dim3((Dn * Dn + 255) / 256), blk, 0, stream>>>(Wo, wot, Dn, Dn);
    transpose_bf16_kernel<<<dim3((Dn * Fn + 255) / 256), blk, 0, stream>>>(W1, w1t, Dn, Fn);
    transpose_bf16_kernel<<<dim3((Fn * Dn + 255) / 256), blk, 0, stream>>>(W2, w2t, Fn, Dn);
    attn_bias_kernel<<<dim3((Bn * Sn + 255) / 256), blk, 0, stream>>>(msk, gw, attb, Bn * Sn);

    // 2) QKV projections: block tile 128x64 -> 512 blocks each
    gemm_bf16_kernel<<<g512, blk, 0, stream>>>(xbf, wqt, bq, nullptr, qbuf, Mtot, Dn, Dn, 2);
    gemm_bf16_kernel<<<g512, blk, 0, stream>>>(xbf, wkt, bk, nullptr, kbuf, Mtot, Dn, Dn, 3);
    gemm_bf16_kernel<<<g512, blk, 0, stream>>>(xbf, wvt, bv, nullptr, vtbf, Mtot, Dn, Dn, 4);

    // 3) flash attention: B*H*8 = 512 blocks (8 q-tiles per block)
    attn_kernel<<<g512, blk, 0, stream>>>(qbuf, kbuf, vtbf, attb, ctxb);

    // 4) output projection -> f32
    gemm_bf16_kernel<<<g512, blk, 0, stream>>>(ctxb, wot, bo, sapre, nullptr, Mtot, Dn, Dn, 0);

    // 5) residual + LN1 (f32 + bf16 copy for the FFN A-matrix)
    add_ln_kernel<<<dim3(Mtot), blk, 0, stream>>>(sapre, x, ln1g, ln1b, saf, sabf);

    // 6) FFN1 + exact GELU -> bf16
    gemm_bf16_kernel<<<g2048, blk, 0, stream>>>(sabf, w1t, b1, nullptr, hbf, Mtot, Fn, Dn, 1);

    // 7) FFN2 -> f32
    gemm_bf16_kernel<<<g512, blk, 0, stream>>>(hbf, w2t, b2, ffnf, nullptr, Mtot, Dn, Fn, 0);

    // 8) residual + LN2 -> d_out
    add_ln_kernel<<<dim3(Mtot), blk, 0, stream>>>(ffnf, saf, ln2g, ln2b, out, nullptr);
}
